// SparseAttention_46712064311931
// MI455X (gfx1250) — matrix operand order repaired and checked by reading, not compiled, not run on hardware
//
#include <hip/hip_runtime.h>
#include <hip/hip_bf16.h>

// Problem constants (from reference)
#define D_MODEL 1024
#define ATTN    64
#define TOPK    32
#define SEQLEN  4096
#define NBATCH  2
#define NROWS   (NBATCH * SEQLEN)   // 8192

typedef __attribute__((ext_vector_type(16))) __bf16        v16bf;
typedef __attribute__((ext_vector_type(8)))  float         v8f;
typedef __attribute__((ext_vector_type(4)))  unsigned int  u32x4;

union Frag16 { u32x4 u[2]; v16bf v; };

static __device__ __forceinline__ v16bf make_frag(u32x4 a, u32x4 b) {
  Frag16 f; f.u[0] = a; f.u[1] = b; return f.v;
}

// ---------------------------------------------------------------------------
// Elementwise f32 -> bf16
// ---------------------------------------------------------------------------
__global__ void f32_to_bf16(const float* __restrict__ in,
                            __hip_bfloat16* __restrict__ out, size_t n) {
  size_t i = (size_t)blockIdx.x * blockDim.x + threadIdx.x;
  if (i < n) out[i] = __float2bfloat16(in[i]);
}

// ---------------------------------------------------------------------------
// Transpose + convert: W (K x N, f32, row-major) -> Wt (N x K, bf16)
// ---------------------------------------------------------------------------
__global__ void transpose_f32_to_bf16(const float* __restrict__ W,
                                      __hip_bfloat16* __restrict__ Wt,
                                      int K, int N) {
  size_t idx = (size_t)blockIdx.x * blockDim.x + threadIdx.x;
  if (idx >= (size_t)K * N) return;
  int k = (int)(idx / N);
  int n = (int)(idx % N);
  Wt[(size_t)n * K + k] = __float2bfloat16(W[idx]);
}

// ---------------------------------------------------------------------------
// WMMA bf16 GEMM:  C[M x N] = (A[M x Kd] @ Bt[N x Kd]^T + bias) * oscale
// Per-wave tile: 32 (M) x 64 (N) = 8 accumulators; A fragments reused x4.
// 8 waves stacked along M -> workgroup tile 256(M) x 64(N).
// ---------------------------------------------------------------------------
template <bool OUT_BF16>
__global__ __launch_bounds__(256)
void wmma_gemm_bf16(const __hip_bfloat16* __restrict__ A,
                    const __hip_bfloat16* __restrict__ Bt,
                    const float* __restrict__ bias,
                    void* __restrict__ Cout,
                    int M, int N, int Kd, float oscale) {
  const int wave = threadIdx.x >> 5;
  const int lane = threadIdx.x & 31;
  const int m    = lane & 15;
  const int hi   = lane >> 4;

  const int m0 = blockIdx.y * 256 + wave * 32;   // this wave: rows m0..m0+31
  const int n0 = blockIdx.x * 64;                // cols n0..n0+63

  const u32x4* __restrict__ Arow0 = (const u32x4*)(A + (size_t)(m0 + m) * Kd);
  const u32x4* __restrict__ Arow1 = (const u32x4*)(A + (size_t)(m0 + 16 + m) * Kd);
  const u32x4* __restrict__ Brow0 = (const u32x4*)(Bt + (size_t)(n0 + m) * Kd);
  const u32x4* __restrict__ Brow1 = (const u32x4*)(Bt + (size_t)(n0 + 16 + m) * Kd);
  const u32x4* __restrict__ Brow2 = (const u32x4*)(Bt + (size_t)(n0 + 32 + m) * Kd);
  const u32x4* __restrict__ Brow3 = (const u32x4*)(Bt + (size_t)(n0 + 48 + m) * Kd);

  v8f c[2][4];
#pragma unroll
  for (int t = 0; t < 4; ++t) {
    const float bval = bias[n0 + 16 * t + m];
    const v8f binit = { bval, bval, bval, bval, bval, bval, bval, bval };
    c[0][t] = binit;
    c[1][t] = binit;
  }

  const int aoff = hi * 8;
  const int boff = hi * 8;

  for (int kk = 0; kk < Kd; kk += 32) {
    const int ia0 = (kk + aoff) >> 3;
    const int ia1 = (kk + 16 + aoff) >> 3;
    const v16bf a0 = make_frag(Arow0[ia0], Arow0[ia1]);
    const v16bf a1 = make_frag(Arow1[ia0], Arow1[ia1]);

    const int ib = (kk + boff) >> 3;
    const u32x4* __restrict__ Brows[4] = { Brow0, Brow1, Brow2, Brow3 };
#pragma unroll
    for (int t = 0; t < 4; ++t) {
      const v16bf b = make_frag(Brows[t][ib], Brows[t][ib + 2]);
      c[0][t] = __builtin_amdgcn_wmma_f32_16x16x32_bf16(false, a0, false, b,
                                                        (short)0, c[0][t], false, false);
      c[1][t] = __builtin_amdgcn_wmma_f32_16x16x32_bf16(false, a1, false, b,
                                                        (short)0, c[1][t], false, false);
    }
  }

#pragma unroll
  for (int i = 0; i < 2; ++i) {
#pragma unroll
    for (int t = 0; t < 4; ++t) {
#pragma unroll
      for (int r = 0; r < 8; ++r) {
        const int row = m0 + 16 * i + hi * 8 + r;
        const int col = n0 + 16 * t + m;
        const float v = c[i][t][r] * oscale;
        if (OUT_BF16)
          ((__hip_bfloat16*)Cout)[(size_t)row * N + col] = __float2bfloat16(v);
        else
          ((float*)Cout)[(size_t)row * N + col] = v;
      }
    }
  }
}

// ---------------------------------------------------------------------------
// WMMA scores + running top-32, with async double-buffered K staging.
// One WG (128 thr = 4 waves) handles 16 query rows. The next 64-key K chunk
// (8 KB, contiguous in Kbf) is streamed into LDS with
// GLOBAL_LOAD_ASYNC_TO_LDS_B128 (ASYNCcnt) while the current chunk's WMMAs
// and the top-k merge run. B fragments are then read from LDS with
// ds_load_b128; +16B row padding (row stride 36 dwords) spreads the 16
// fragment lanes across distinct banks.
// ---------------------------------------------------------------------------
#define KROW_U16 72                     // 64 data + 8 pad (bf16 elements)
#define KBUF_U16 (64 * KROW_U16)        // one 64-key chunk, 9216 B

static __device__ __forceinline__ void async_copy_chunk(const unsigned short* gsrc,
                                                        unsigned ldsdst, int tid) {
  // 8 KB = 512 x 16B units; 128 threads x 4 units, coalesced in global.
#pragma unroll
  for (int u = 0; u < 4; ++u) {
    const int g   = tid + u * 128;      // 16B unit index
    const int row = g >> 3;             // 8 units per 128B key row
    const int col = g & 7;
    const unsigned lds = ldsdst + (unsigned)(row * (KROW_U16 * 2) + col * 16);
    const void* gp = (const void*)((const char*)gsrc + (size_t)g * 16);
    asm volatile("global_load_async_to_lds_b128 %0, %1, off"
                 :: "v"(lds), "v"(gp) : "memory");
  }
}

__global__ __launch_bounds__(128)
void topk_wmma(const __hip_bfloat16* __restrict__ Qbf,
               const __hip_bfloat16* __restrict__ Kbf,
               int* __restrict__ topi, float* __restrict__ topp) {
  __shared__ unsigned short Bs[2][KBUF_U16];   // double-buffered K chunk (18.4 KB)
  __shared__ float chunk[16][68];              // 16 rows x 64 keys, bank-spread
  __shared__ float lvals[16][TOPK];
  __shared__ int   lidx[16][TOPK];

  const int tid  = threadIdx.x;
  const int wave = tid >> 5;
  const int lane = tid & 31;
  const int m    = lane & 15;
  const int hi   = lane >> 4;

  const int row0  = blockIdx.x * 16;   // 16 query rows
  const int batch = row0 >> 12;        // row0 / SEQLEN

  // Q fragments for this 16-row tile are loop-invariant (K-dim = 64).
  const u32x4* __restrict__ Arow = (const u32x4*)(Qbf + (size_t)(row0 + m) * ATTN);
  const v16bf alo = make_frag(Arow[hi],     Arow[2 + hi]);   // K = 0..31
  const v16bf ahi = make_frag(Arow[4 + hi], Arow[6 + hi]);   // K = 32..63

  // init top-k lists
  {
    float* lv = &lvals[0][0];
    int*   li = &lidx[0][0];
    for (int j = tid; j < 16 * TOPK; j += 128) { lv[j] = -__builtin_inff(); li[j] = 0; }
  }

  const __hip_bfloat16* __restrict__ Kb = Kbf + (size_t)batch * SEQLEN * ATTN;
  const unsigned ldsB0 = (unsigned)(size_t)&Bs[0][0];   // low 32 bits = LDS offset
  const unsigned ldsB1 = (unsigned)(size_t)&Bs[1][0];

  // Preload chunk 0 into buffer 0.
  async_copy_chunk((const unsigned short*)Kb, ldsB0, tid);
  asm volatile("s_wait_asynccnt 0" ::: "memory");
  __syncthreads();

  int buf = 0;
  for (int c0 = 0; c0 < SEQLEN; c0 += 64) {
    // Kick off the next chunk while this one computes/merges.
    if (c0 + 64 < SEQLEN)
      async_copy_chunk((const unsigned short*)(Kb + (size_t)(c0 + 64) * ATTN),
                       buf ? ldsB0 : ldsB1, tid);

    // B fragments from LDS (ds_load_b128): key row = wave*16 + m.
    const u32x4* __restrict__ bp =
        (const u32x4*)&Bs[buf][(wave * 16 + m) * KROW_U16];
    const v16bf blo = make_frag(bp[hi],     bp[2 + hi]);   // K = 0..31
    const v16bf bhi = make_frag(bp[4 + hi], bp[6 + hi]);   // K = 32..63

    v8f s = { 0.f, 0.f, 0.f, 0.f, 0.f, 0.f, 0.f, 0.f };
    s = __builtin_amdgcn_wmma_f32_16x16x32_bf16(false, alo, false, blo, (short)0, s, false, false);
    s = __builtin_amdgcn_wmma_f32_16x16x32_bf16(false, ahi, false, bhi, (short)0, s, false, false);

#pragma unroll
    for (int r = 0; r < 8; ++r)
      chunk[r + 8 * hi][wave * 16 + m] = s[r];
    __syncthreads();

    if (tid < 16) {
      float* __restrict__ lv = lvals[tid];
      int*   __restrict__ li = lidx[tid];
      float  vmin = lv[TOPK - 1];
#pragma clang loop unroll(disable)
      for (int j = 0; j < 64; ++j) {
        const float v = chunk[tid][j];
        if (v > vmin) {
          int p = TOPK - 1;
#pragma clang loop unroll(disable)
          while (p > 0 && lv[p - 1] < v) { lv[p] = lv[p - 1]; li[p] = li[p - 1]; --p; }
          lv[p] = v; li[p] = c0 + j;
          vmin = lv[TOPK - 1];
        }
      }
    }
    // Ensure our async copies landed, then rendezvous before switching buffers.
    asm volatile("s_wait_asynccnt 0" ::: "memory");
    __syncthreads();
    buf ^= 1;
  }

  // softmax over each row's top-32 and emit (index, prob)
  if (tid < 16) {
    const int row = row0 + tid;
    const float* lv = lvals[tid];
    const int*   li = lidx[tid];
    const float mx = lv[0];
    float s = 0.f;
    for (int j = 0; j < TOPK; ++j) s += __expf(lv[j] - mx);
    const float inv = 1.f / s;
    for (int j = 0; j < TOPK; ++j) {
      topi[(size_t)row * TOPK + j] = li[j];
      topp[(size_t)row * TOPK + j] = __expf(lv[j] - mx) * inv;
    }
  }
}

// ---------------------------------------------------------------------------
// Sparse attn @ V: out_row = sum_j p_j * V[idx_j, :]  (32 gathered rows).
// ---------------------------------------------------------------------------
__global__ __launch_bounds__(256)
void gather_av(const int* __restrict__ topi, const float* __restrict__ topp,
               const __hip_bfloat16* __restrict__ V,
               __hip_bfloat16* __restrict__ AO) {
  __shared__ int   si[TOPK];
  __shared__ float sp[TOPK];
  const int row   = blockIdx.x;
  const int batch = row >> 12;
  const int tid   = threadIdx.x;
  if (tid < TOPK) {
    si[tid] = topi[(size_t)row * TOPK + tid];
    sp[tid] = topp[(size_t)row * TOPK + tid];
  }
  __syncthreads();

  const int d0 = tid * 4;
  float a0 = 0.f, a1 = 0.f, a2 = 0.f, a3 = 0.f;
  for (int j = 0; j < TOPK; ++j) {
    const float p = sp[j];
    const size_t off = (size_t)(batch * SEQLEN + si[j]) * D_MODEL + d0;
    if (j + 1 < TOPK) {
      const size_t noff = (size_t)(batch * SEQLEN + si[j + 1]) * D_MODEL + d0;
      __builtin_prefetch((const void*)(V + noff), 0, 1);   // global_prefetch_b8
    }
    const unsigned int* vr = (const unsigned int*)(V + off);  // 2 bf16 per u32
    const unsigned int w0 = vr[0], w1 = vr[1];
    a0 += p * __uint_as_float(w0 << 16);
    a1 += p * __uint_as_float(w0 & 0xffff0000u);
    a2 += p * __uint_as_float(w1 << 16);
    a3 += p * __uint_as_float(w1 & 0xffff0000u);
  }
  __hip_bfloat16* o = AO + (size_t)row * D_MODEL + d0;
  o[0] = __float2bfloat16(a0);
  o[1] = __float2bfloat16(a1);
  o[2] = __float2bfloat16(a2);
  o[3] = __float2bfloat16(a3);
}

// ---------------------------------------------------------------------------
// Host-side pipeline
// ---------------------------------------------------------------------------
extern "C" void kernel_launch(void* const* d_in, const int* in_sizes, int n_in,
                              void* d_out, int out_size, void* d_ws, size_t ws_size,
                              hipStream_t stream) {
  const float* x  = (const float*)d_in[0];
  const float* Wq = (const float*)d_in[1];
  const float* bq = (const float*)d_in[2];
  const float* Wk = (const float*)d_in[3];
  const float* bk = (const float*)d_in[4];
  const float* Wv = (const float*)d_in[5];
  const float* bv = (const float*)d_in[6];
  const float* Wo = (const float*)d_in[7];
  const float* bo = (const float*)d_in[8];
  float* out = (float*)d_out;
  (void)in_sizes; (void)n_in; (void)out_size; (void)ws_size;

  char* ws = (char*)d_ws;
  size_t off = 0;
  auto alloc = [&](size_t bytes) -> char* {
    char* p = ws + off;
    off += (bytes + 255) & ~(size_t)255;
    return p;
  };

  __hip_bfloat16* xbf  = (__hip_bfloat16*)alloc((size_t)NROWS * D_MODEL * 2);
  __hip_bfloat16* WqT  = (__hip_bfloat16*)alloc((size_t)ATTN * D_MODEL * 2);
  __hip_bfloat16* WkT  = (__hip_bfloat16*)alloc((size_t)ATTN * D_MODEL * 2);
  __hip_bfloat16* WvT  = (__hip_bfloat16*)alloc((size_t)D_MODEL * D_MODEL * 2);
  __hip_bfloat16* WoT  = (__hip_bfloat16*)alloc((size_t)D_MODEL * D_MODEL * 2);
  __hip_bfloat16* Qbf  = (__hip_bfloat16*)alloc((size_t)NROWS * ATTN * 2);
  __hip_bfloat16* Kbf  = (__hip_bfloat16*)alloc((size_t)NROWS * ATTN * 2);
  __hip_bfloat16* Vbf  = (__hip_bfloat16*)alloc((size_t)NROWS * D_MODEL * 2);
  __hip_bfloat16* AObf = (__hip_bfloat16*)alloc((size_t)NROWS * D_MODEL * 2);
  int*            topi = (int*)alloc((size_t)NROWS * TOPK * 4);
  float*          topp = (float*)alloc((size_t)NROWS * TOPK * 4);

  // 1) bf16 conversion of activations and transposed weights
  {
    const size_t n = (size_t)NROWS * D_MODEL;
    f32_to_bf16<<<dim3((unsigned)((n + 255) / 256)), 256, 0, stream>>>(x, xbf, n);
  }
  {
    const size_t nQK = (size_t)D_MODEL * ATTN;
    transpose_f32_to_bf16<<<dim3((unsigned)((nQK + 255) / 256)), 256, 0, stream>>>(Wq, WqT, D_MODEL, ATTN);
    transpose_f32_to_bf16<<<dim3((unsigned)((nQK + 255) / 256)), 256, 0, stream>>>(Wk, WkT, D_MODEL, ATTN);
    const size_t nVO = (size_t)D_MODEL * D_MODEL;
    transpose_f32_to_bf16<<<dim3((unsigned)((nVO + 255) / 256)), 256, 0, stream>>>(Wv, WvT, D_MODEL, D_MODEL);
    transpose_f32_to_bf16<<<dim3((unsigned)((nVO + 255) / 256)), 256, 0, stream>>>(Wo, WoT, D_MODEL, D_MODEL);
  }

  // 2) Projections via WMMA. Q carries the 1/sqrt(64) score scale (exact
  //    power of two, folded before the bf16 round).
  dim3 gQK(ATTN / 64, NROWS / 256);     // (1, 32)
  wmma_gemm_bf16<true><<<gQK, 256, 0, stream>>>(xbf, WqT, bq, Qbf, NROWS, ATTN, D_MODEL, 0.125f);
  wmma_gemm_bf16<true><<<gQK, 256, 0, stream>>>(xbf, WkT, bk, Kbf, NROWS, ATTN, D_MODEL, 1.0f);
  dim3 gV(D_MODEL / 64, NROWS / 256);   // (16, 32)
  wmma_gemm_bf16<true><<<gV, 256, 0, stream>>>(xbf, WvT, bv, Vbf, NROWS, D_MODEL, D_MODEL, 1.0f);

  // 3) WMMA scores + top-32 + softmax (16 query rows per WG)
  topk_wmma<<<dim3(NROWS / 16), 128, 0, stream>>>(Qbf, Kbf, topi, topp);

  // 4) Sparse attn @ V gather
  gather_av<<<dim3(NROWS), 256, 0, stream>>>(topi, topp, Vbf, AObf);

  // 5) Output projection via WMMA -> f32 result
  wmma_gemm_bf16<false><<<gV, 256, 0, stream>>>(AObf, WoT, bo, out, NROWS, D_MODEL, D_MODEL, 1.0f);
}
